// DRL4TSP_11682311045556
// MI455X (gfx1250) — compile-verified, matrix operand order checked by reading, not run on hardware
//
#include <hip/hip_runtime.h>
#include <hip/hip_bf16.h>
#include <math.h>

// ---------------------------------------------------------------------------
// DRL4TSP pointer-network greedy decode for MI455X (gfx1250, wave32, WMMA).
// B=128 batch rows decode independently -> 8 blocks x 16 rows (WMMA M=16).
// Step-invariant attention matmuls are hoisted into precompute kernels.
// Per-step GEMMs (GRU gates, attn query, pointer term) run on
// v_wmma_f32_16x16x32_f16 with f32 accumulation and bias folded into C.
// ---------------------------------------------------------------------------

typedef _Float16 half_t;
typedef __attribute__((ext_vector_type(16))) _Float16 v16h;
typedef __attribute__((ext_vector_type(8)))  float    v8f;

constexpr int kB = 128;   // batch
constexpr int kS = 64;    // sequence (cities)
constexpr int kH = 128;   // hidden
constexpr int kH3 = 384;  // 3*H
constexpr int FBLK = kB * kH * kS;  // 1,048,576 floats per [B][H][S] block

// workspace layout (floats):
//   [0      , FBLK)   static_h  f32  [B][H][S]
//   [FBLK   , 2FBLK)  dynamic_h f32  [B][H][S]
//   [2FBLK  , 3FBLK)  eAT       f32  [B][S][H]  (attn_W[:, :2H] @ [sh;dh], transposed)
//   [3FBLK  , 4FBLK)  pAT       f32  [B][S][H]  (ptr_W[:, :H] @ sh, transposed)
// then halfs at (float*)ws + 4*FBLK:
//   Wih (3H*H) | Whh (3H*H) | W3 (H*H) | P2 (H*H)   = 131072 halfs
constexpr int kWihOff = 0;
constexpr int kWhhOff = kH3 * kH;            // 49152
constexpr int kW3Off  = 2 * kH3 * kH;        // 98304
constexpr int kP2Off  = 2 * kH3 * kH + kH * kH; // 114688
constexpr int kHalfTotal = kP2Off + kH * kH; // 131072

// ---------------------------------------------------------------------------
// WMMA helpers (CDNA5 16x16x32 f16 -> f32)
// ---------------------------------------------------------------------------
__device__ inline v8f wmma16(v16h a, v16h b, v8f c) {
  return __builtin_amdgcn_wmma_f32_16x16x32_f16(
      /*neg_a=*/false, a, /*neg_b=*/false, b,
      /*c_mod=*/(short)0, c, /*reuse_a=*/false, /*reuse_b=*/false);
}

// 16-bit operand fragment, row-major source (row = matrix row for A with
// row index = lane&15; for B we feed W^T by indexing the weight row n0+n).
// ISA layout: lanes 0-15 carry K = k0..k0+7 and k0+16..k0+23 (packed pairs),
// lanes 16-31 carry K = k0+8..k0+15 and k0+24..k0+31.
__device__ inline v16h frag_rm(const half_t* __restrict__ base, int stride,
                               int row, int k0, int lane) {
  const int kb = (lane < 16) ? 0 : 8;
  const half_t* p = base + row * stride + k0 + kb;
  v16h a;
#pragma unroll
  for (int i = 0; i < 8; ++i) a[i] = p[i];
#pragma unroll
  for (int i = 0; i < 8; ++i) a[8 + i] = p[16 + i];
  return a;
}

// f32 C/D layout: VGPR r holds M=r (lanes 0-15) / M=8+r (lanes 16-31), N=lane&15
__device__ inline void store_acc(float* __restrict__ dst, int stride, int n0,
                                 int lane, v8f acc) {
  const int n = (lane & 15) + n0;
  const int mb = (lane < 16) ? 0 : 8;
#pragma unroll
  for (int r = 0; r < 8; ++r) dst[(mb + r) * stride + n] = acc[r];
}

__device__ inline v8f splat8(float v) {
  v8f a;
#pragma unroll
  for (int r = 0; r < 8; ++r) a[r] = v;
  return a;
}

// ---------------------------------------------------------------------------
// Precompute kernels
// ---------------------------------------------------------------------------
__global__ void k_prep_weights(const float* __restrict__ wih,
                               const float* __restrict__ whh,
                               const float* __restrict__ aW,
                               const float* __restrict__ pW,
                               half_t* __restrict__ hb) {
  for (int i = blockIdx.x * blockDim.x + threadIdx.x; i < kHalfTotal;
       i += gridDim.x * blockDim.x) {
    float v;
    if (i < kWhhOff) {
      v = wih[i];
    } else if (i < kW3Off) {
      v = whh[i - kWhhOff];
    } else if (i < kP2Off) {
      int j = i - kW3Off;
      int h = j >> 7, k = j & 127;
      v = aW[h * kH3 + 2 * kH + k];   // attn_W[:, 2H:3H]
    } else {
      int j = i - kP2Off;
      int h = j >> 7, k = j & 127;
      v = pW[h * (2 * kH) + kH + k];  // ptr_W[:, H:2H]
    }
    hb[i] = (half_t)v;
  }
}

__global__ void k_encode(const float* __restrict__ stat,
                         const float* __restrict__ dyn,
                         const float* __restrict__ sw, const float* __restrict__ sb,
                         const float* __restrict__ dw, const float* __restrict__ db,
                         float* __restrict__ ws) {
  for (int i = blockIdx.x * blockDim.x + threadIdx.x; i < 2 * FBLK;
       i += gridDim.x * blockDim.x) {
    if (i < FBLK) {
      int b = i / (kH * kS), r = i % (kH * kS), h = r / kS, s = r % kS;
      ws[i] = sb[h] + sw[h * 2] * stat[b * 128 + s] +
              sw[h * 2 + 1] * stat[b * 128 + 64 + s];
    } else {
      int j = i - FBLK;
      int b = j / (kH * kS), r = j % (kH * kS), h = r / kS, s = r % kS;
      ws[i] = db[h] + dw[h * 2] * dyn[b * 128 + s] +
              dw[h * 2 + 1] * dyn[b * 128 + 64 + s];
    }
  }
}

__global__ void k_prep_attn(const float* __restrict__ aW,
                            const float* __restrict__ pW,
                            float* __restrict__ ws) {
  const float* sh_ = ws;          // static_h  [B][H][S]
  const float* dh_ = ws + FBLK;   // dynamic_h [B][H][S]
  for (int i = blockIdx.x * blockDim.x + threadIdx.x; i < 2 * FBLK;
       i += gridDim.x * blockDim.x) {
    if (i < FBLK) {               // eAT[b][s][h]
      int b = i >> 13, r = i & 8191, s = r >> 7, h = r & 127;
      const float* sp = sh_ + (size_t)b * 8192 + s;
      const float* dp = dh_ + (size_t)b * 8192 + s;
      const float* w0 = aW + h * kH3;
      float acc = 0.f;
#pragma unroll 4
      for (int k = 0; k < kH; ++k)
        acc += w0[k] * sp[k * kS] + w0[kH + k] * dp[k * kS];
      ws[2 * FBLK + i] = acc;
    } else {                      // pAT[b][s][h]
      int j = i - FBLK;
      int b = j >> 13, r = j & 8191, s = r >> 7, h = r & 127;
      const float* sp = sh_ + (size_t)b * 8192 + s;
      const float* w0 = pW + h * (2 * kH);
      float acc = 0.f;
#pragma unroll 4
      for (int k = 0; k < kH; ++k) acc += w0[k] * sp[k * kS];
      ws[3 * FBLK + j] = acc;
    }
  }
}

// ---------------------------------------------------------------------------
// Sequential decode: 8 blocks x 128 threads (4 waves), 16 batch rows/block.
// ---------------------------------------------------------------------------
__global__ __launch_bounds__(128) void k_decode(
    const float* __restrict__ stat, const float* __restrict__ x0,
    const float* __restrict__ dec_w, const float* __restrict__ dec_b,
    const float* __restrict__ bih, const float* __restrict__ bhh,
    const float* __restrict__ av, const float* __restrict__ pv,
    const float* __restrict__ ws, const half_t* __restrict__ wih_h,
    const half_t* __restrict__ whh_h, const half_t* __restrict__ w3_h,
    const half_t* __restrict__ p2_h, float* __restrict__ out) {
  __shared__ float  sh_h[16 * 128];    // hidden state f32
  __shared__ half_t sh_x[16 * 128];    // dec_h (GRU input) f16  [A operand]
  __shared__ half_t sh_hf[16 * 128];   // hidden f16             [A operand]
  __shared__ half_t sh_c[16 * 128];    // context f16            [A operand]
  __shared__ float  sh_grz[16 * 256];  // fused r,z preacts; later q | c2
  __shared__ float  sh_in[16 * 128];   // i_n preact; later attns [16][64]
  __shared__ float  sh_hn[16 * 128];   // h_n preact
  __shared__ float  sh_din[16 * 2];    // decoder input (2 channels)

  const int tid = threadIdx.x;
  const int lane = tid & 31;
  const int wv = tid >> 5;             // wave id 0..3
  const int b0 = blockIdx.x * 16;
  const float* eAT = ws + 2 * FBLK;
  const float* pAT = ws + 3 * FBLK;
  const float* sth = ws;

  // Warm L2 with the f16 weights (global_prefetch_b8).
  for (int i = tid * 64; i < kHalfTotal * 2; i += 128 * 64)
    __builtin_prefetch((const char*)wih_h + i, 0, 1);

  for (int i = tid; i < 16 * 128; i += 128) {
    sh_h[i] = 0.f;
    sh_hf[i] = (half_t)0.f;
  }
  if (tid < 32) sh_din[tid] = x0[tid & 1];
  __syncthreads();

  for (int t = 0; t < kS; ++t) {
    // ---- 1. dec_h = decoder_w @ dec_in + decoder_b (K=2) -> f16 A-staging
    for (int i = tid; i < 2048; i += 128) {
      int m = i >> 7, h = i & 127;
      float v = dec_b[h] + dec_w[h * 2] * sh_din[m * 2] +
                dec_w[h * 2 + 1] * sh_din[m * 2 + 1];
      sh_x[i] = (half_t)v;
    }
    __syncthreads();

    // ---- 2. GRU gate GEMMs via WMMA, bias in C, gi chained after gh.
    // r,z tiles: N in [0,256), 16 tiles, 4 per wave; acc = (x@Wih^T + h@Whh^T + b)
#pragma unroll
    for (int it = 0; it < 4; ++it) {
      const int n0 = (wv * 4 + it) * 16;
      const int nn = n0 + (lane & 15);
      v8f acc = splat8(bih[nn] + bhh[nn]);
#pragma unroll
      for (int k0 = 0; k0 < 128; k0 += 32)
        acc = wmma16(frag_rm(sh_hf, 128, lane & 15, k0, lane),
                     frag_rm(whh_h, 128, nn, k0, lane), acc);
#pragma unroll
      for (int k0 = 0; k0 < 128; k0 += 32)
        acc = wmma16(frag_rm(sh_x, 128, lane & 15, k0, lane),
                     frag_rm(wih_h, 128, nn, k0, lane), acc);
      store_acc(sh_grz, 256, n0, lane, acc);
    }
    // n-gate tiles: N in [256,384): keep i_n and h_n separate (r gates h_n)
#pragma unroll
    for (int it = 0; it < 2; ++it) {
      const int j = wv * 2 + it;       // 0..7
      const int n0 = 256 + j * 16;     // weight row block
      const int c0 = j * 16;           // local column block
      const int nn = n0 + (lane & 15);
      v8f ah = splat8(bhh[nn]);
      v8f ai = splat8(bih[nn]);
#pragma unroll
      for (int k0 = 0; k0 < 128; k0 += 32)
        ah = wmma16(frag_rm(sh_hf, 128, lane & 15, k0, lane),
                    frag_rm(whh_h, 128, nn, k0, lane), ah);
#pragma unroll
      for (int k0 = 0; k0 < 128; k0 += 32)
        ai = wmma16(frag_rm(sh_x, 128, lane & 15, k0, lane),
                    frag_rm(wih_h, 128, nn, k0, lane), ai);
      store_acc(sh_hn, 128, c0, lane, ah);
      store_acc(sh_in, 128, c0, lane, ai);
    }
    __syncthreads();

    // ---- 3. GRU gates elementwise; h_new -> sh_h (f32) + sh_hf (f16)
    for (int i = tid; i < 2048; i += 128) {
      int m = i >> 7, h = i & 127;
      float rr = 1.f / (1.f + expf(-sh_grz[m * 256 + h]));
      float zz = 1.f / (1.f + expf(-sh_grz[m * 256 + 128 + h]));
      float nn = tanhf(sh_in[i] + rr * sh_hn[i]);
      float hn = (1.f - zz) * nn + zz * sh_h[i];
      sh_h[i] = hn;
      sh_hf[i] = (half_t)hn;
    }
    __syncthreads();

    // ---- 4. q = h_new @ W3^T  (16x128x128) -> sh_grz cols [0,128)
#pragma unroll
    for (int it = 0; it < 2; ++it) {
      const int n0 = (wv * 2 + it) * 16;
      const int nn = n0 + (lane & 15);
      v8f acc = splat8(0.f);
#pragma unroll
      for (int k0 = 0; k0 < 128; k0 += 32)
        acc = wmma16(frag_rm(sh_hf, 128, lane & 15, k0, lane),
                     frag_rm(w3_h, 128, nn, k0, lane), acc);
      store_acc(sh_grz, 256, n0, lane, acc);
    }
    __syncthreads();

    // ---- 5. attention scores e[b,s] = sum_h av[h]*tanh(eAT+q), softmax over s
    {
      const int m = tid >> 3, u = tid & 7;
      const int b = b0 + m;
      float sc[8];
#pragma unroll
      for (int si = 0; si < 8; ++si) {
        const int s = si * 8 + u;
        const float* ea = eAT + (size_t)b * 8192 + s * 128;
        const float* q = sh_grz + m * 256;
        float e = 0.f;
#pragma unroll 4
        for (int h = 0; h < 128; ++h) e += av[h] * tanhf(ea[h] + q[h]);
        sc[si] = e;
      }
      float mx = sc[0];
#pragma unroll
      for (int si = 1; si < 8; ++si) mx = fmaxf(mx, sc[si]);
#pragma unroll
      for (int msk = 1; msk < 8; msk <<= 1)
        mx = fmaxf(mx, __shfl_xor(mx, msk, 32));
      float ex[8], sum = 0.f;
#pragma unroll
      for (int si = 0; si < 8; ++si) { ex[si] = expf(sc[si] - mx); sum += ex[si]; }
#pragma unroll
      for (int msk = 1; msk < 8; msk <<= 1) sum += __shfl_xor(sum, msk, 32);
      const float inv = 1.f / sum;
#pragma unroll
      for (int si = 0; si < 8; ++si)
        sh_in[m * 128 + si * 8 + u] = ex[si] * inv;  // attns in sh_in[.., 0..63]
    }
    __syncthreads();

    // ---- 6. context[m][h] = sum_s attns * static_h -> f16 A operand
    for (int i = tid; i < 2048; i += 128) {
      int m = i >> 7, h = i & 127;
      const float* sp = sth + (size_t)(b0 + m) * 8192 + h * 64;
      const float* at = sh_in + m * 128;
      float c = 0.f;
#pragma unroll 4
      for (int s = 0; s < 64; ++s) c += at[s] * sp[s];
      sh_c[i] = (half_t)c;
    }
    __syncthreads();

    // ---- 7. c2 = context @ P2^T -> sh_grz cols [128,256)
#pragma unroll
    for (int it = 0; it < 2; ++it) {
      const int n0 = (wv * 2 + it) * 16;
      const int nn = n0 + (lane & 15);
      v8f acc = splat8(0.f);
#pragma unroll
      for (int k0 = 0; k0 < 128; k0 += 32)
        acc = wmma16(frag_rm(sh_c, 128, lane & 15, k0, lane),
                     frag_rm(p2_h, 128, nn, k0, lane), acc);
      store_acc(sh_grz, 256, 128 + n0, lane, acc);
    }
    __syncthreads();

    // ---- 8. pointer logits, greedy argmax + log-softmax, outputs
    {
      const int m = tid >> 3, u = tid & 7;
      const int b = b0 + m;
      float lg[8];
#pragma unroll
      for (int si = 0; si < 8; ++si) {
        const int s = si * 8 + u;
        const float* pa = pAT + (size_t)b * 8192 + s * 128;
        const float* c2 = sh_grz + m * 256 + 128;
        float l = 0.f;
#pragma unroll 4
        for (int h = 0; h < 128; ++h) l += pv[h] * tanhf(pa[h] + c2[h]);
        lg[si] = l;
      }
      // argmax (first max wins, matching jnp.argmax)
      float bv = lg[0];
      int bs = u;
#pragma unroll
      for (int si = 1; si < 8; ++si) {
        const int s = si * 8 + u;
        if (lg[si] > bv) { bv = lg[si]; bs = s; }
      }
#pragma unroll
      for (int msk = 1; msk < 8; msk <<= 1) {
        float ov = __shfl_xor(bv, msk, 32);
        int os = __shfl_xor(bs, msk, 32);
        if (ov > bv || (ov == bv && os < bs)) { bv = ov; bs = os; }
      }
      // log-softmax at argmax: logp = -log(sum exp(l - max))
      float sm = 0.f;
#pragma unroll
      for (int si = 0; si < 8; ++si) sm += expf(lg[si] - bv);
#pragma unroll
      for (int msk = 1; msk < 8; msk <<= 1) sm += __shfl_xor(sm, msk, 32);
      if (u == 0) {
        out[b * kS + t] = (float)bs;                       // tour_idx
        out[kB * kS + b * kS + t] = -logf(sm);             // tour_logp
        sh_din[m * 2 + 0] = stat[b * 128 + bs];            // next dec input
        sh_din[m * 2 + 1] = stat[b * 128 + 64 + bs];
      }
    }
    __syncthreads();
  }
}

// ---------------------------------------------------------------------------
extern "C" void kernel_launch(void* const* d_in, const int* in_sizes, int n_in,
                              void* d_out, int out_size, void* d_ws, size_t ws_size,
                              hipStream_t stream) {
  const float* stat  = (const float*)d_in[0];
  const float* dyn   = (const float*)d_in[1];
  const float* x0    = (const float*)d_in[2];
  const float* sw    = (const float*)d_in[3];
  const float* sb    = (const float*)d_in[4];
  const float* dw    = (const float*)d_in[5];
  const float* db    = (const float*)d_in[6];
  const float* decw  = (const float*)d_in[7];
  const float* decb  = (const float*)d_in[8];
  const float* wih   = (const float*)d_in[9];
  const float* whh   = (const float*)d_in[10];
  const float* gbih  = (const float*)d_in[11];
  const float* gbhh  = (const float*)d_in[12];
  const float* av    = (const float*)d_in[13];
  const float* aW    = (const float*)d_in[14];
  const float* pv    = (const float*)d_in[15];
  const float* pW    = (const float*)d_in[16];

  float* ws = (float*)d_ws;
  half_t* hb = (half_t*)(ws + 4 * FBLK);

  k_prep_weights<<<512, 256, 0, stream>>>(wih, whh, aW, pW, hb);
  k_encode<<<4096, 256, 0, stream>>>(stat, dyn, sw, sb, dw, db, ws);
  k_prep_attn<<<4096, 256, 0, stream>>>(aW, pW, ws);
  k_decode<<<8, 128, 0, stream>>>(stat, x0, decw, decb, gbih, gbhh, av, pv, ws,
                                  hb + kWihOff, hb + kWhhOff, hb + kW3Off,
                                  hb + kP2Off, (float*)d_out);
}
